// SparseAttention1D_19774029430910
// MI455X (gfx1250) — matrix operand order repaired and checked by reading, not compiled, not run on hardware
//
#include <hip/hip_runtime.h>
#include <hip/hip_bf16.h>

// ---------------------------------------------------------------------------
// CDNA5 (gfx1250) fused windowed-attention pipeline, bf16 WMMA everywhere.
//   Stage 1: qkv = w_qkv @ x            (f32 -> bf16 convert-on-stage GEMM)
//   Stage 2: flash-style local attention per 512-wide window (bf16 WMMA,
//            f32 accumulate, online softmax)
//   Stage 3: y = w_out @ attn_out + b   (bf16-in / f32-out GEMM)
// Round 3: 16B-vectorized global loads in attention staging (kills the
//          >256-VGPR pressure / s_set_vgpr_msb), LDS-transposed epilogues so
//          all global stores are b128, register-pipelined GEMM staging.
// ---------------------------------------------------------------------------

typedef __attribute__((ext_vector_type(16))) __bf16 v16bf;
typedef __attribute__((ext_vector_type(8)))  __bf16 v8bf;
typedef __attribute__((ext_vector_type(8)))  float  v8f;
typedef __attribute__((ext_vector_type(4)))  float  v4f;

#define N_CTX     32768
#define DIMC      256
#define HIDDEN    512
#define HEADS     8
#define DIM_HEAD  64
#define WINDOW    512
#define ATT_SCALE 0.125f   // 64^-0.5

template <bool C, class A, class B> struct cond_t            { using T = A; };
template <class A, class B>        struct cond_t<false, A, B>{ using T = B; };

__device__ inline __bf16 to_bf16(float v)  { return (__bf16)v; }
__device__ inline float  to_f32(float v)   { return v; }
__device__ inline float  to_f32(__bf16 v)  { return (float)v; }

__device__ inline v8f vzero8() {
    v8f z;
#pragma unroll
    for (int i = 0; i < 8; ++i) z[i] = 0.0f;
    return z;
}

// A-matrix fragment 16x32 bf16 (contiguous K, matches load_B_frag):
//   lanes 0-15 : row M = lane,    elems 0..15 -> K 0..15
//   lanes 16-31: row M = lane-16, elems 0..15 -> K 16..31
__device__ inline v16bf load_A_frag(const __bf16* base, int stride, int lane) {
    const int r = lane & 15, half = lane >> 4;
    const __bf16* p = base + r * stride;
    union { v16bf v; v8bf h[2]; } u;
    u.h[0] = *(const v8bf*)(p + 16 * half);
    u.h[1] = *(const v8bf*)(p + 16 * half + 8);
    return u.v;
}

// B-matrix fragment 32x16 bf16 (lanes 0-15: K=0..15 of column N=lane,
// lanes 16-31: K=16..31). LDS tile stored n-major: Bs[n][k].
__device__ inline v16bf load_B_frag(const __bf16* base, int stride, int lane) {
    const int c = lane & 15, half = lane >> 4;
    const __bf16* p = base + c * stride + 16 * half;
    union { v16bf v; v8bf h[2]; } u;
    u.h[0] = *(const v8bf*)(p);
    u.h[1] = *(const v8bf*)(p + 8);
    return u.v;
}

__device__ inline v8f wmma_bf16(v16bf a, v16bf b, v8f c) {
    return __builtin_amdgcn_wmma_f32_16x16x32_bf16(
        false, a, false, b, (short)0, c, false, false);
}

// ---------------------------------------------------------------------------
// Tiled GEMM: C[z] = W(MxK,f32) * B[z](KxN) (+bias), bf16 WMMA core.
// Block = 256 threads (8 waves), tile 128(M) x 128(N), K step 32.
// Wave grid 4(M) x 2(N); each wave owns 32x64 -> 8 C tiles, 8 WMMA / K-step.
// Register-staged pipelining: tile k+1 global loads issue before tile k WMMAs.
// Epilogue transposes through LDS so every global store is 16B.
// ---------------------------------------------------------------------------
template <typename TB, bool OUT_BF16, bool HAS_BIAS>
__global__ __launch_bounds__(256) void gemm_wmma_kernel(
    const float* __restrict__ Wp, const TB* __restrict__ Bbase,
    void* __restrict__ Cbase, const float* __restrict__ bias,
    int K, int N, size_t strideB, size_t strideC) {

    constexpr int BM = 128, BN = 128, BK = 32, LDT = 40;  // 80B rows, 16B aligned
    using EpiT = typename cond_t<OUT_BF16, __bf16, float>::T;
    constexpr int LDE = 136;                              // epi row stride (16B aligned)
    constexpr size_t STAGE_B = (size_t)(BM + BN) * LDT * 2;
    constexpr size_t EPI_B   = (size_t)BM * LDE * sizeof(EpiT);
    constexpr size_t SMEM_B  = EPI_B > STAGE_B ? EPI_B : STAGE_B;
    __shared__ __align__(16) char smem[SMEM_B];
    __bf16 (*As)[LDT] = (__bf16(*)[LDT])smem;                       // [m][k]
    __bf16 (*Bs)[LDT] = (__bf16(*)[LDT])(smem + (size_t)BM * LDT * 2); // [n][k]
    EpiT  (*Ct)[LDE]  = (EpiT(*)[LDE])smem;                         // epilogue

    const int t    = threadIdx.x;
    const int lane = t & 31;
    const int wid  = t >> 5;
    const int bm0  = blockIdx.y * BM;
    const int bn0  = blockIdx.x * BN;
    const int wm   = (wid & 3) * 32;   // wave M offset inside block tile
    const int wn   = (wid >> 2) * 64;  // wave N offset inside block tile

    const TB* Bp = Bbase + (size_t)blockIdx.z * strideB;

    // Per-thread staging slots: A row am cols [ak0,ak0+16); B row bk cols [bnn,+16).
    const int am = t >> 1, ak0 = (t & 1) * 16;
    const int bk = t >> 3, bnn = (t & 7) * 16;

    v8f acc[2][4];
#pragma unroll
    for (int i = 0; i < 2; ++i)
#pragma unroll
        for (int j = 0; j < 4; ++j) acc[i][j] = vzero8();

    float ra[16], rb[16];
#pragma unroll
    for (int i = 0; i < 16; ++i)
        ra[i] = Wp[(size_t)(bm0 + am) * K + (ak0 + i)];
#pragma unroll
    for (int i = 0; i < 16; ++i)
        rb[i] = to_f32(Bp[(size_t)bk * N + (bn0 + bnn + i)]);

    for (int k0 = 0; k0 < K; k0 += BK) {
        __syncthreads();  // previous compute done reading LDS
#pragma unroll
        for (int i = 0; i < 16; ++i) As[am][ak0 + i] = to_bf16(ra[i]);
#pragma unroll
        for (int i = 0; i < 16; ++i) Bs[bnn + i][bk] = to_bf16(rb[i]);
        __syncthreads();

        if (k0 + BK < K) {  // next tile's loads retire during the WMMAs below
            const int kn = k0 + BK;
#pragma unroll
            for (int i = 0; i < 16; ++i)
                ra[i] = Wp[(size_t)(bm0 + am) * K + (kn + ak0 + i)];
#pragma unroll
            for (int i = 0; i < 16; ++i)
                rb[i] = to_f32(Bp[(size_t)(kn + bk) * N + (bn0 + bnn + i)]);
        }

        v16bf a0 = load_A_frag(&As[wm][0],      LDT, lane);
        v16bf a1 = load_A_frag(&As[wm + 16][0], LDT, lane);
#pragma unroll
        for (int j = 0; j < 4; ++j) {
            v16bf bj = load_B_frag(&Bs[wn + 16 * j][0], LDT, lane);
            acc[0][j] = wmma_bf16(a0, bj, acc[0][j]);
            acc[1][j] = wmma_bf16(a1, bj, acc[1][j]);
        }
    }

    // Epilogue: C tile layout (VGPR v, lane L -> M = v+8*(L>>4), N = L&15)
    // -> LDS transpose buffer -> 16B global stores.
    __syncthreads();  // all waves done reading As/Bs
    const int nn = lane & 15, mb = 8 * (lane >> 4);
#pragma unroll
    for (int i = 0; i < 2; ++i) {
#pragma unroll
        for (int j = 0; j < 4; ++j) {
#pragma unroll
            for (int v = 0; v < 8; ++v) {
                const int ml = wm + 16 * i + mb + v;
                float val = acc[i][j][v];
                if constexpr (HAS_BIAS) val += bias[bm0 + ml];
                Ct[ml][wn + 16 * j + nn] = (EpiT)val;
            }
        }
    }
    __syncthreads();

    EpiT* C = (EpiT*)Cbase + (size_t)blockIdx.z * strideC;
    constexpr int VE = OUT_BF16 ? 8 : 4;        // elems per 16B store
    constexpr int CPT = (BM * BN) / (VE * 256); // chunks per thread
    using VecT = typename cond_t<OUT_BF16, v8bf, v4f>::T;
#pragma unroll
    for (int r = 0; r < CPT; ++r) {
        const int c = t + 256 * r;
        const int m = c / (BN / VE);
        const int nc = (c % (BN / VE)) * VE;
        *(VecT*)(C + (size_t)(bm0 + m) * N + (bn0 + nc)) =
            *(const VecT*)(&Ct[m][nc]);
    }
}

// ---------------------------------------------------------------------------
// Stage 2: flash-style local attention. One 128-thread block (4 waves) per
// (batch, head, window, 64-row i-block). Online softmax over 8 j-blocks.
// All global traffic is 16B-vectorized.
// ---------------------------------------------------------------------------
__global__ __launch_bounds__(128) void attn_window_kernel(
    const __bf16* __restrict__ qkv, __bf16* __restrict__ aout) {

    constexpr int LD = 72;  // 144B row stride (16B aligned), bank-spread
    __shared__ __bf16 Qt[64][LD];  // [i][d] : A of S-GEMM
    __shared__ __bf16 Kt[64][LD];  // [j][d] : B of S-GEMM; reused as O-transpose
    __shared__ __bf16 Vt[64][LD];  // [d][j] : A of O-GEMM
    __shared__ __bf16 Pt[64][LD];  // [i][j] : B of O-GEMM (n-major)
    __shared__ float m_s[64], l_s[64], a_s[64];

    const int t    = threadIdx.x;
    const int lane = t & 31;
    const int w    = t >> 5;
    const int half = lane >> 4;
    const int lc   = lane & 15;

    const int bh  = blockIdx.z;           // b*8 + h
    const int b   = bh >> 3, h = bh & 7;
    const int win = blockIdx.y;
    const int i0  = blockIdx.x * 64;
    const int p0  = win * WINDOW;

    const size_t nb = (size_t)b * (3 * HIDDEN) * N_CTX;
    const __bf16* qp = qkv + nb + (size_t)(h * DIM_HEAD) * N_CTX;
    const __bf16* kp = qkv + nb + (size_t)(HIDDEN + h * DIM_HEAD) * N_CTX;
    const __bf16* vp = qkv + nb + (size_t)(2 * HIDDEN + h * DIM_HEAD) * N_CTX;

    // Per-thread staging chunk map: 512 16B-chunks over [64 d][8 pos-groups].
    const int cd0 = t >> 3;          // d row for chunk r=0 (advance by 16 per r)
    const int cj  = (t & 7) * 8;     // position-group offset

    // Q staging: v8bf global loads, scaled, transposed into Qt[i][d].
#pragma unroll
    for (int r = 0; r < 4; ++r) {
        const int d = cd0 + 16 * r;
        v8bf qv = *(const v8bf*)(qp + (size_t)d * N_CTX + p0 + i0 + cj);
#pragma unroll
        for (int e = 0; e < 8; ++e)
            Qt[cj + e][d] = (__bf16)((float)qv[e] * ATT_SCALE);
    }
    if (t < 64) { m_s[t] = -1e30f; l_s[t] = 0.0f; }
    __syncthreads();

    const int iw = 16 * w;  // this wave's i-strip (S) and d-strip (O)

    // Q fragments are loop-invariant: hoist out of the j-loop.
    const v16bf aq0 = load_A_frag(&Qt[iw][0],  LD, lane);
    const v16bf aq1 = load_A_frag(&Qt[iw][32], LD, lane);

    v8f o[4];
#pragma unroll
    for (int it = 0; it < 4; ++it) o[it] = vzero8();

    for (int jb = 0; jb < 8; ++jb) {
        const int j0 = jb * 64;
        // K/V staging: v8bf global loads; K transposed, V direct (b128 ds store).
#pragma unroll
        for (int r = 0; r < 4; ++r) {
            const int d = cd0 + 16 * r;
            const size_t g = (size_t)d * N_CTX + p0 + j0 + cj;
            v8bf kv = *(const v8bf*)(kp + g);
            v8bf vv = *(const v8bf*)(vp + g);
#pragma unroll
            for (int e = 0; e < 8; ++e) Kt[cj + e][d] = kv[e];
            *(v8bf*)(&Vt[d][cj]) = vv;
        }
        __syncthreads();

        // S = (Q^T)K : M=i, N=j, K=d(64) in two 32-chunks
        v8f s[4];
#pragma unroll
        for (int jt = 0; jt < 4; ++jt) {
            v16bf bk0 = load_B_frag(&Kt[16 * jt][0],  LD, lane);
            v16bf bk1 = load_B_frag(&Kt[16 * jt][32], LD, lane);
            v8f c = wmma_bf16(aq0, bk0, vzero8());
            s[jt]   = wmma_bf16(aq1, bk1, c);
        }

        // Online softmax over this wave's 16 rows (iw + 8*half + v).
#pragma unroll
        for (int v = 0; v < 8; ++v) {
            float rm = s[0][v];
#pragma unroll
            for (int jt = 1; jt < 4; ++jt) rm = fmaxf(rm, s[jt][v]);
#pragma unroll
            for (int off = 8; off > 0; off >>= 1)
                rm = fmaxf(rm, __shfl_xor(rm, off, 32));  // stays in 16-lane half

            const int row = iw + 8 * half + v;
            const float mo = m_s[row];
            const float mn = fmaxf(mo, rm);
            const float al = __expf(mo - mn);

            float rs = 0.0f;
#pragma unroll
            for (int jt = 0; jt < 4; ++jt) {
                float p = __expf(s[jt][v] - mn);
                s[jt][v] = p;
                rs += p;
            }
#pragma unroll
            for (int off = 8; off > 0; off >>= 1)
                rs += __shfl_xor(rs, off, 32);

            if (lc == 0) {
                m_s[row] = mn;
                l_s[row] = al * l_s[row] + rs;
                a_s[row] = al;
            }
#pragma unroll
            for (int jt = 0; jt < 4; ++jt)
                Pt[row][16 * jt + lc] = (__bf16)s[jt][v];
        }
        __syncthreads();  // all P tiles + alphas visible

        // O += V * P^T : M=d, N=i, K=j(64). Rescale O by alpha (per-lane i).
        v16bf av0 = load_A_frag(&Vt[iw][0],  LD, lane);
        v16bf av1 = load_A_frag(&Vt[iw][32], LD, lane);
#pragma unroll
        for (int it = 0; it < 4; ++it) {
            const float al = a_s[16 * it + lc];
            o[it] = o[it] * al;
            v16bf bp0 = load_B_frag(&Pt[16 * it][0],  LD, lane);
            v16bf bp1 = load_B_frag(&Pt[16 * it][32], LD, lane);
            o[it] = wmma_bf16(av0, bp0, o[it]);
            o[it] = wmma_bf16(av1, bp1, o[it]);
        }
        __syncthreads();  // done reading Kt/Vt/Pt before next staging pass
    }

    // Epilogue: O /= l, transpose through LDS (reuse Kt as Ot[d][i]),
    // then 16B global stores (channel rows contiguous over position).
#pragma unroll
    for (int it = 0; it < 4; ++it) {
        const int iloc = 16 * it + lc;
        const float linv = 1.0f / l_s[iloc];
#pragma unroll
        for (int v = 0; v < 8; ++v) {
            const int d = iw + 8 * half + v;
            Kt[d][iloc] = (__bf16)(o[it][v] * linv);
        }
    }
    __syncthreads();

    __bf16* op = aout + ((size_t)b * HIDDEN + h * DIM_HEAD) * N_CTX;
#pragma unroll
    for (int r = 0; r < 4; ++r) {
        const int d = cd0 + 16 * r;
        *(v8bf*)(op + (size_t)d * N_CTX + p0 + i0 + cj) =
            *(const v8bf*)(&Kt[d][cj]);
    }
}

// ---------------------------------------------------------------------------
extern "C" void kernel_launch(void* const* d_in, const int* in_sizes, int n_in,
                              void* d_out, int out_size, void* d_ws, size_t ws_size,
                              hipStream_t stream) {
    (void)in_sizes; (void)n_in; (void)out_size; (void)ws_size;
    const float* x     = (const float*)d_in[0];  // [2, 256, 32768]
    const float* w_qkv = (const float*)d_in[1];  // [1536, 256]
    const float* w_out = (const float*)d_in[2];  // [256, 512]
    const float* b_out = (const float*)d_in[3];  // [256]
    float* y = (float*)d_out;                    // [2, 256, 32768]

    const int B = 2;
    __bf16* qkv  = (__bf16*)d_ws;                                  // 192 MB
    __bf16* aout = (__bf16*)((char*)d_ws +
                   (size_t)B * 3 * HIDDEN * N_CTX * sizeof(__bf16)); // +64 MB

    // Stage 1: qkv[z] = w_qkv @ x[z]  (M=1536, N=32768, K=256) -> bf16
    {
        dim3 grid(N_CTX / 128, (3 * HIDDEN) / 128, B);
        gemm_wmma_kernel<float, true, false><<<grid, 256, 0, stream>>>(
            w_qkv, x, qkv, nullptr, DIMC, N_CTX,
            (size_t)DIMC * N_CTX, (size_t)3 * HIDDEN * N_CTX);
    }
    // Stage 2: windowed attention (8 i-blocks x 64 windows x b*h)
    {
        dim3 grid(WINDOW / 64, N_CTX / WINDOW, B * HEADS);
        attn_window_kernel<<<grid, 128, 0, stream>>>(qkv, aout);
    }
    // Stage 3: y[z] = w_out @ aout[z] + b_out  (M=256, N=32768, K=512) -> f32
    {
        dim3 grid(N_CTX / 128, DIMC / 128, B);
        gemm_wmma_kernel<__bf16, false, true><<<grid, 256, 0, stream>>>(
            w_out, aout, y, b_out, HIDDEN, N_CTX,
            (size_t)HIDDEN * N_CTX, (size_t)DIMC * N_CTX);
    }
}
